// EnhancedHydraulicGNNv3_79551384256835
// MI455X (gfx1250) — matrix operand order repaired and checked by reading, not compiled, not run on hardware
//
#include <hip/hip_runtime.h>
#include <hip/hip_bf16.h>
#include <math.h>

#define NN 50000
#define EE 800000
#define MM (EE + NN)
#define CC 128
#define BB 64

typedef _Float16 v16h __attribute__((ext_vector_type(16)));
typedef _Float16 v8h  __attribute__((ext_vector_type(8)));
typedef float    v8f  __attribute__((ext_vector_type(8)));

// ---------- helpers ----------
__device__ __forceinline__ unsigned fenc(float f) {
    unsigned b = __float_as_uint(f);
    unsigned m = (b & 0x80000000u) ? 0xFFFFFFFFu : 0x80000000u;
    return b ^ m;
}
__device__ __forceinline__ float fdec(unsigned u) {
    unsigned m = (u & 0x80000000u) ? 0x80000000u : 0xFFFFFFFFu;
    return __uint_as_float(u ^ m);
}
__device__ __forceinline__ float sigm(float x) { return 1.f / (1.f + __expf(-x)); }

// ---------- weight f32 -> f16 pack into WMMA B-fragment order ----------
// For W[K,128] (zero-padded to KPAD): element (k,n) goes to
//   t = n>>4, c = k>>5, hi = (k>>4)&1, h = k&15, lane = hi*16 + (n&15)
//   dst[ ((t*(KPAD/32) + c)*32 + lane)*16 + h ]
// so each wave's B fragment for (tile t, chunk c) is one contiguous v16h per lane.
__global__ void pack_w16(const float* __restrict__ src, _Float16* __restrict__ dst,
                         int K, int KPAD) {
    int i = blockIdx.x * blockDim.x + threadIdx.x;
    if (i >= KPAD * CC) return;
    int k = i / CC, n = i % CC;
    float v = (k < K) ? src[k * CC + n] : 0.f;
    int t = n >> 4, c = k >> 5, hi = (k >> 4) & 1, h = k & 15;
    int lane = hi * 16 + (n & 15);
    dst[(((size_t)t * (KPAD / 32) + c) * 32 + lane) * 16 + h] = (_Float16)v;
}

// ---------- per-layer accumulator init ----------
__global__ void init_layer(unsigned* __restrict__ emax, float* __restrict__ den,
                           float* __restrict__ agg) {
    int i = blockIdx.x * blockDim.x + threadIdx.x;
    if (i < NN * CC) agg[i] = 0.f;
    if (i < NN * 4) { emax[i] = 0x007FFFFFu; den[i] = 0.f; }  // enc(-inf)
}

__global__ void init_pool(unsigned* __restrict__ smax, float* __restrict__ sden,
                          float* __restrict__ cnt, float* __restrict__ pooled) {
    int i = blockIdx.x * blockDim.x + threadIdx.x;
    if (i < BB * CC) pooled[i] = 0.f;
    if (i < BB) { smax[i] = 0x007FFFFFu; sden[i] = 0.f; cnt[i] = 0.f; }
}

// ---------- WMMA GEMM: out[N,128] = X[N,K] @ W[K,128] + bias ----------
// blockIdx.x = M-tile (16 rows), blockIdx.y = 0 (Wl) / 1 (Wr). 8 waves, one 16-col tile each.
template <int KPAD>
__global__ void __launch_bounds__(256) gemm_wmma(
    const float* __restrict__ X, int K,
    const _Float16* __restrict__ Wl16, const _Float16* __restrict__ Wr16,
    const float* __restrict__ bl, const float* __restrict__ br,
    float* __restrict__ outL, float* __restrict__ outR) {
    __shared__ __align__(16) _Float16 As[16 * KPAD];
    const int row0 = blockIdx.x * 16;
    const _Float16* W = (blockIdx.y == 0) ? Wl16 : Wr16;
    const float* bias = (blockIdx.y == 0) ? bl : br;
    float* out = (blockIdx.y == 0) ? outL : outR;

    // stage A tile (f32 -> f16, zero-pad K..KPAD)
    if (K == KPAD) {
        // vectorized path: float4 loads, 4-half stores
        const float4* X4 = (const float4*)X;
        for (int i = threadIdx.x; i < 16 * (KPAD / 4); i += 256) {
            int r = i / (KPAD / 4), q = i % (KPAD / 4);
            float4 v = X4[(size_t)(row0 + r) * (K / 4) + q];
            _Float16* p = &As[r * KPAD + q * 4];
            p[0] = (_Float16)v.x; p[1] = (_Float16)v.y;
            p[2] = (_Float16)v.z; p[3] = (_Float16)v.w;
        }
    } else {
        for (int i = threadIdx.x; i < 16 * KPAD; i += 256) {
            int r = i / KPAD, k = i % KPAD;
            float v = (k < K) ? X[(size_t)(row0 + r) * K + k] : 0.f;
            As[r * KPAD + k] = (_Float16)v;
        }
    }
    __syncthreads();

    const int wave = threadIdx.x >> 5;
    const int lane = threadIdx.x & 31;
    const int mrow = lane & 15;
    const int hi = (lane >> 4) & 1;

    const v8h* Arow = (const v8h*)(As + mrow * KPAD);
    const v16h* Wp = (const v16h*)W;

    v8f acc = {};
#pragma unroll
    for (int c = 0; c < KPAD / 32; c++) {
        // A fragment: k = c*32 + hi*16 + 0..7 and c*32 + hi*16 + 8 + 0..7
        v8h alo = Arow[c * 4 + 2 * hi];
        v8h ahi = Arow[c * 4 + 2 * hi + 1];
        v16h a;
#pragma unroll
        for (int h = 0; h < 8; h++) { a[h] = alo[h]; a[h + 8] = ahi[h]; }
        // B fragment: pre-packed, contiguous per lane
        v16h bf = Wp[((size_t)wave * (KPAD / 32) + c) * 32 + lane];
        acc = __builtin_amdgcn_wmma_f32_16x16x32_f16(false, a, false, bf, (short)0, acc,
                                                     false, false);
    }

    const int col = wave * 16 + mrow;
    const float bv = bias[col];
#pragma unroll
    for (int r = 0; r < 8; r++) {
        int row = row0 + r + hi * 8;
        out[(size_t)row * CC + col] = acc[r] + bv;
    }
}

// ---------- edge pass 1: attention scores + segment max ----------
__global__ void __launch_bounds__(256) edge_pass1(
    const int* __restrict__ ei, const float* __restrict__ hl, const float* __restrict__ hr,
    const float* __restrict__ att, float* __restrict__ eb, unsigned* __restrict__ emax) {
    int m = blockIdx.x * 8 + (threadIdx.x >> 5);
    if (m >= MM) return;
    int lane = threadIdx.x & 31;
    int s, d;
    if (m < EE) { s = ei[m]; d = ei[EE + m]; } else { s = d = m - EE; }

    float4 a = *(const float4*)(hl + (size_t)s * CC + lane * 4);
    float4 b = *(const float4*)(hr + (size_t)d * CC + lane * 4);
    float4 at = *(const float4*)(att + lane * 4);
    float v0 = a.x + b.x, v1 = a.y + b.y, v2 = a.z + b.z, v3 = a.w + b.w;
    v0 = v0 > 0.f ? v0 : 0.2f * v0;
    v1 = v1 > 0.f ? v1 : 0.2f * v1;
    v2 = v2 > 0.f ? v2 : 0.2f * v2;
    v3 = v3 > 0.f ? v3 : 0.2f * v3;
    float p = v0 * at.x + v1 * at.y + v2 * at.z + v3 * at.w;
    p += __shfl_xor(p, 1);
    p += __shfl_xor(p, 2);
    p += __shfl_xor(p, 4);
    int head = lane >> 3;
    if ((lane & 7) == 0) {
        eb[m * 4 + head] = p;
        atomicMax(&emax[d * 4 + head], fenc(p));
    }
}

// ---------- edge pass 2: exp weights, denominator, weighted aggregation ----------
__global__ void __launch_bounds__(256) edge_pass2(
    const int* __restrict__ ei, const float* __restrict__ hl, const float* __restrict__ eb,
    const unsigned* __restrict__ emax, float* __restrict__ den, float* __restrict__ agg) {
    int m = blockIdx.x * 8 + (threadIdx.x >> 5);
    if (m >= MM) return;
    int lane = threadIdx.x & 31;
    int s, d;
    if (m < EE) { s = ei[m]; d = ei[EE + m]; } else { s = d = m - EE; }
    int head = lane >> 3;
    float ew = __expf(eb[m * 4 + head] - fdec(emax[d * 4 + head]));
    if ((lane & 7) == 0) atomicAdd(&den[d * 4 + head], ew);
    float4 a = *(const float4*)(hl + (size_t)s * CC + lane * 4);
    float* ap = agg + (size_t)d * CC + lane * 4;
    atomicAdd(ap + 0, ew * a.x);
    atomicAdd(ap + 1, ew * a.y);
    atomicAdd(ap + 2, ew * a.z);
    atomicAdd(ap + 3, ew * a.w);
}

// ---------- node finalize: normalize, bias, layernorm(128), ELU ----------
__global__ void __launch_bounds__(256) node_finalize(
    const float* __restrict__ agg, const float* __restrict__ den,
    const float* __restrict__ bias, const float* __restrict__ lng,
    const float* __restrict__ lnb, float* __restrict__ hout) {
    int n = blockIdx.x * 8 + (threadIdx.x >> 5);
    if (n >= NN) return;
    int lane = threadIdx.x & 31;
    int c0 = lane * 4;
    int head = lane >> 3;
    float dn = den[n * 4 + head];
    float4 a = *(const float4*)(agg + (size_t)n * CC + c0);
    float4 bv = *(const float4*)(bias + c0);
    float v[4] = {a.x / dn + bv.x, a.y / dn + bv.y, a.z / dn + bv.z, a.w / dn + bv.w};
    float s = v[0] + v[1] + v[2] + v[3];
    for (int o = 16; o >= 1; o >>= 1) s += __shfl_xor(s, o);
    float mu = s * (1.f / 128.f);
    float q = 0.f;
#pragma unroll
    for (int j = 0; j < 4; j++) { float dd = v[j] - mu; q += dd * dd; }
    for (int o = 16; o >= 1; o >>= 1) q += __shfl_xor(q, o);
    float rs = rsqrtf(q * (1.f / 128.f) + 1e-5f);
    float4 g = *(const float4*)(lng + c0);
    float4 b = *(const float4*)(lnb + c0);
    float o0 = (v[0] - mu) * rs * g.x + b.x;
    float o1 = (v[1] - mu) * rs * g.y + b.y;
    float o2 = (v[2] - mu) * rs * g.z + b.z;
    float o3 = (v[3] - mu) * rs * g.w + b.w;
    o0 = o0 > 0.f ? o0 : __expf(o0) - 1.f;
    o1 = o1 > 0.f ? o1 : __expf(o1) - 1.f;
    o2 = o2 > 0.f ? o2 : __expf(o2) - 1.f;
    o3 = o3 > 0.f ? o3 : __expf(o3) - 1.f;
    float4 r = {o0, o1, o2, o3};
    *(float4*)(hout + (size_t)n * CC + c0) = r;
}

// ---------- pooling scores ----------
__global__ void __launch_bounds__(256) pool_score(
    const float* __restrict__ h, const float* __restrict__ w1, const float* __restrict__ b1,
    const float* __restrict__ w2, const float* __restrict__ b2, const int* __restrict__ batch,
    float* __restrict__ sbuf, unsigned* __restrict__ smax) {
    int n = blockIdx.x * 8 + (threadIdx.x >> 5);
    if (n >= NN) return;
    int lane = threadIdx.x & 31;
    float a0 = b1[lane], a1 = b1[lane + 32];
    const float* hr = h + (size_t)n * CC;
    for (int c = 0; c < CC; c++) {
        float xv = hr[c];
        a0 += xv * w1[c * 64 + lane];
        a1 += xv * w1[c * 64 + lane + 32];
    }
    float p = tanhf(a0) * w2[lane] + tanhf(a1) * w2[lane + 32];
    for (int o = 16; o >= 1; o >>= 1) p += __shfl_xor(p, o);
    if (lane == 0) {
        float sv = p + b2[0];
        sbuf[n] = sv;
        atomicMax(&smax[batch[n]], fenc(sv));
    }
}

// ---------- pooling accumulation ----------
__global__ void __launch_bounds__(256) pool_accum(
    const float* __restrict__ h, const int* __restrict__ batch, const float* __restrict__ sbuf,
    const unsigned* __restrict__ smax, float* __restrict__ sden, float* __restrict__ cnt,
    float* __restrict__ pooled) {
    int n = blockIdx.x * 8 + (threadIdx.x >> 5);
    if (n >= NN) return;
    int lane = threadIdx.x & 31;
    int b = batch[n];
    float ew = __expf(sbuf[n] - fdec(smax[b]));
    if (lane == 0) { atomicAdd(&sden[b], ew); atomicAdd(&cnt[b], 1.f); }
    float4 hv = *(const float4*)(h + (size_t)n * CC + lane * 4);
    float* pp = pooled + (size_t)b * CC + lane * 4;
    atomicAdd(pp + 0, ew * hv.x);
    atomicAdd(pp + 1, ew * hv.y);
    atomicAdd(pp + 2, ew * hv.z);
    atomicAdd(pp + 3, ew * hv.w);
}

// ---------- LSTM + classifier head (B=64, one thread per row) ----------
__device__ __forceinline__ void lnorm(float* v, int n, const float* g, const float* b) {
    float mu = 0.f;
    for (int k = 0; k < n; k++) mu += v[k];
    mu /= n;
    float var = 0.f;
    for (int k = 0; k < n; k++) { float d = v[k] - mu; var += d * d; }
    var /= n;
    float rs = rsqrtf(var + 1e-5f);
    for (int k = 0; k < n; k++) v[k] = (v[k] - mu) * rs * g[k] + b[k];
}

__global__ void head_kernel(
    const float* __restrict__ pooled, const float* __restrict__ sden, const float* __restrict__ cnt,
    const float* wih0, const float* bih0, const float* bhh0,
    const float* wih1, const float* bih1, const float* bhh1,
    const float* cw1, const float* cb1, const float* g1, const float* b1,
    const float* rw1, const float* rb1, const float* rg1, const float* rbb1,
    const float* rw2, const float* rb2, const float* rg2, const float* rbb2,
    const float* cw2, const float* cb2, const float* g2, const float* b2,
    const float* cw3, const float* cb3, float* __restrict__ out) {
    int b = threadIdx.x;
    if (b >= BB) return;
    float den = sden[b];
    float cn = fmaxf(cnt[b], 1.f);
    float g[128];
    for (int c = 0; c < 128; c++) g[c] = pooled[b * 128 + c] / den / cn;

    // LSTM layer 0 (h0=c0=0 => whh and forget gate are dead)
    float h1[32];
    for (int j = 0; j < 32; j++) {
        float gi = bih0[j] + bhh0[j];
        float gg = bih0[64 + j] + bhh0[64 + j];
        float go = bih0[96 + j] + bhh0[96 + j];
        for (int c = 0; c < 128; c++) {
            float v = g[c];
            gi += v * wih0[j * 128 + c];
            gg += v * wih0[(64 + j) * 128 + c];
            go += v * wih0[(96 + j) * 128 + c];
        }
        float cc = sigm(gi) * tanhf(gg);
        h1[j] = sigm(go) * tanhf(cc);
    }
    // LSTM layer 1
    float t[32];
    for (int j = 0; j < 32; j++) {
        float gi = bih1[j] + bhh1[j];
        float gg = bih1[64 + j] + bhh1[64 + j];
        float go = bih1[96 + j] + bhh1[96 + j];
        for (int c = 0; c < 32; c++) {
            float v = h1[c];
            gi += v * wih1[j * 32 + c];
            gg += v * wih1[(64 + j) * 32 + c];
            go += v * wih1[(96 + j) * 32 + c];
        }
        float cc = sigm(gi) * tanhf(gg);
        t[j] = sigm(go) * tanhf(cc);
    }
    // z = relu(ln(t @ cw1 + cb1))
    float z[32];
    for (int j = 0; j < 32; j++) {
        float a = cb1[j];
        for (int k = 0; k < 32; k++) a += t[k] * cw1[k * 32 + j];
        z[j] = a;
    }
    lnorm(z, 32, g1, b1);
    for (int j = 0; j < 32; j++) z[j] = fmaxf(z[j], 0.f);
    // r = relu(ln(z @ rw1 + rb1))
    float r[32];
    for (int j = 0; j < 32; j++) {
        float a = rb1[j];
        for (int k = 0; k < 32; k++) a += z[k] * rw1[k * 32 + j];
        r[j] = a;
    }
    lnorm(r, 32, rg1, rbb1);
    for (int j = 0; j < 32; j++) r[j] = fmaxf(r[j], 0.f);
    // r2 = ln(r @ rw2 + rb2)
    float r2[32];
    for (int j = 0; j < 32; j++) {
        float a = rb2[j];
        for (int k = 0; k < 32; k++) a += r[k] * rw2[k * 32 + j];
        r2[j] = a;
    }
    lnorm(r2, 32, rg2, rbb2);
    for (int j = 0; j < 32; j++) z[j] = fmaxf(r2[j] + z[j], 0.f);
    // z2 = relu(ln(z @ cw2 + cb2))  [16]
    float z2[16];
    for (int j = 0; j < 16; j++) {
        float a = cb2[j];
        for (int k = 0; k < 32; k++) a += z[k] * cw2[k * 16 + j];
        z2[j] = a;
    }
    lnorm(z2, 16, g2, b2);
    for (int j = 0; j < 16; j++) z2[j] = fmaxf(z2[j], 0.f);
    // out = z2 @ cw3 + cb3 [8]
    for (int j = 0; j < 8; j++) {
        float a = cb3[j];
        for (int k = 0; k < 16; k++) a += z2[k] * cw3[k * 8 + j];
        out[b * 8 + j] = a;
    }
}

// ---------- host ----------
extern "C" void kernel_launch(void* const* d_in, const int* in_sizes, int n_in,
                              void* d_out, int out_size, void* d_ws, size_t ws_size,
                              hipStream_t stream) {
    (void)in_sizes; (void)n_in; (void)out_size; (void)ws_size;
    const float* x = (const float*)d_in[0];
    const int* ei = (const int*)d_in[1];
    const int* batch = (const int*)d_in[2];
    auto P = [&](int i) { return (const float*)d_in[i]; };

    char* w = (char*)d_ws;
    size_t o = 0;
    float* Hb  = (float*)(w + o); o += (size_t)NN * CC * 4;
    float* HL  = (float*)(w + o); o += (size_t)NN * CC * 4;
    float* HR  = (float*)(w + o); o += (size_t)NN * CC * 4;
    float* AGG = (float*)(w + o); o += (size_t)NN * CC * 4;
    float* EB  = (float*)(w + o); o += (size_t)MM * 4 * 4;
    unsigned* EMX = (unsigned*)(w + o); o += (size_t)NN * 4 * 4;
    float* DEN = (float*)(w + o); o += (size_t)NN * 4 * 4;
    float* SB  = (float*)(w + o); o += (size_t)NN * 4;
    unsigned* SMX = (unsigned*)(w + o); o += BB * 4;
    float* SDEN = (float*)(w + o); o += BB * 4;
    float* CNTB = (float*)(w + o); o += BB * 4;
    float* POOL = (float*)(w + o); o += BB * CC * 4;
    _Float16* WF = (_Float16*)(w + o);

    const int kdims[3] = {16, 128, 128};
    const int kpads[3] = {32, 128, 128};
    size_t woffL[3], woffR[3];
    {
        size_t cur = 0;
        for (int i = 0; i < 3; i++) {
            woffL[i] = cur; cur += (size_t)kpads[i] * CC;
            woffR[i] = cur; cur += (size_t)kpads[i] * CC;
        }
    }
    for (int i = 0; i < 3; i++) {
        int n = kpads[i] * CC;
        pack_w16<<<(n + 255) / 256, 256, 0, stream>>>(P(3 + 8 * i + 0), WF + woffL[i],
                                                      kdims[i], kpads[i]);
        pack_w16<<<(n + 255) / 256, 256, 0, stream>>>(P(3 + 8 * i + 2), WF + woffR[i],
                                                      kdims[i], kpads[i]);
    }

    const float* hin = x;
    for (int i = 0; i < 3; i++) {
        int base = 3 + 8 * i;
        int K = kdims[i];
        init_layer<<<(NN * CC + 255) / 256, 256, 0, stream>>>(EMX, DEN, AGG);
        if (i == 0)
            gemm_wmma<32><<<dim3(NN / 16, 2), 256, 0, stream>>>(
                hin, K, WF + woffL[i], WF + woffR[i], P(base + 1), P(base + 3), HL, HR);
        else
            gemm_wmma<128><<<dim3(NN / 16, 2), 256, 0, stream>>>(
                hin, K, WF + woffL[i], WF + woffR[i], P(base + 1), P(base + 3), HL, HR);
        edge_pass1<<<(MM + 7) / 8, 256, 0, stream>>>(ei, HL, HR, P(base + 4), EB, EMX);
        edge_pass2<<<(MM + 7) / 8, 256, 0, stream>>>(ei, HL, EB, EMX, DEN, AGG);
        node_finalize<<<(NN + 7) / 8, 256, 0, stream>>>(AGG, DEN, P(base + 5), P(base + 6),
                                                        P(base + 7), Hb);
        hin = Hb;
    }

    init_pool<<<(BB * CC + 255) / 256, 256, 0, stream>>>(SMX, SDEN, CNTB, POOL);
    pool_score<<<(NN + 7) / 8, 256, 0, stream>>>(Hb, P(27), P(28), P(29), P(30), batch, SB, SMX);
    pool_accum<<<(NN + 7) / 8, 256, 0, stream>>>(Hb, batch, SB, SMX, SDEN, CNTB, POOL);

    head_kernel<<<1, BB, 0, stream>>>(
        POOL, SDEN, CNTB,
        P(31), P(33), P(34),       // lstm0 wih, bih, bhh
        P(35), P(37), P(38),       // lstm1 wih, bih, bhh
        P(39), P(40), P(41), P(42),
        P(43), P(44), P(45), P(46),
        P(47), P(48), P(49), P(50),
        P(51), P(52), P(53), P(54),
        P(55), P(56), (float*)d_out);
}
